// ClusterNet_BGC_75642964017167
// MI455X (gfx1250) — matrix operand order repaired and checked by reading, not compiled, not run on hardware
//
#include <hip/hip_runtime.h>
#include <hip/hip_bf16.h>
#include <math.h>

// CDNA5 / gfx1250: wave32, WMMA 16x16x32 f16 -> f32 accumulate.
typedef __attribute__((ext_vector_type(16))) _Float16 v16h;
typedef __attribute__((ext_vector_type(8)))  float    v8f;

// ---------------------------------------------------------------------------
// Implicit-GEMM convolution via V_WMMA_F32_16X16X32_F16.
//   C[M=Cout][N=B*Hout*Wout] = A[M][K] * B[K][N]  (+bias, opt ReLU)
// K is enumerated as (ky, kx, ci) so the inner ci-chunk loop needs no index
// decode: A stride = KH*KW floats, B stride = Hin*Win floats.
// Each wave owns a 16(M) x 64(N) strip: one A fragment feeds 4 WMMAs.
// Fragment layouts used here (contiguous k per lane half; A and B agree):
//   A 16x32: lanes 0-15 -> M=lane, K=e (0..15); lanes 16-31 -> K=16+e
//   B 32x16: lanes 0-15 -> N=lane, K=e (0..15); lanes 16-31 -> K=16+e
//   C f32 : VGPR v -> row v (lanes 0-15) / v+8 (lanes 16-31), col = lane&15
// Modes: 0 = fwd 3x3 stride2 SAME (pad_lo=0); 1 = 1x1 stride1;
//        2 = conv_transpose 3x3 stride2 SAME (pad_a=2, unflipped kernel).
// ---------------------------------------------------------------------------
__global__ void conv_wmma_kernel(const float* __restrict__ in,
                                 const float* __restrict__ w,
                                 const float* __restrict__ bias,
                                 float* __restrict__ out,
                                 int B, int Cin, int Hin, int Win,
                                 int Cout, int Hout, int Wout,
                                 int KH, int KW, int mode, int relu)
{
    const int lane = threadIdx.x & 31;
    const int wave = threadIdx.x >> 5;
    const int Ntot = B * Hout * Wout;
    const int nstrips = (Ntot + 63) >> 6;          // 64 columns per wave
    const int strip = blockIdx.x * 8 + wave;
    const int tileM = blockIdx.y;
    if (strip >= nstrips) return;   // whole-wave exit: EXEC all-ones for WMMA

    const int HWin = Hin * Win;
    const int KHW  = KH * KW;
    const int am   = tileM * 16 + (lane & 15);
    const bool m_ok = (am < Cout);
    const int aKbase = (lane & 16) ? 16 : 0;
    const int bKbase = (lane & 16) ? 16 : 0;
    const float* wrow = w + (size_t)am * Cin * KHW;

    // decode the 4 output columns this lane contributes to (once)
    bool nok[4]; int bbv[4], oyv[4], oxv[4];
    #pragma unroll
    for (int t = 0; t < 4; ++t) {
        int bn = strip * 64 + t * 16 + (lane & 15);
        nok[t] = (bn < Ntot);
        int bb = 0, oy = 0, ox = 0;
        if (nok[t]) {
            bb = bn / (Hout * Wout);
            int r = bn - bb * (Hout * Wout);
            oy = r / Wout;
            ox = r - oy * Wout;
        }
        bbv[t] = bb; oyv[t] = oy; oxv[t] = ox;
    }

    v8f acc[4] = {{}, {}, {}, {}};

    for (int ky = 0; ky < KH; ++ky) {
        for (int kx = 0; kx < KW; ++kx) {
            const int kyx = ky * KW + kx;
            // per-tap validity + base address for each of the 4 columns
            const float* bbase[4]; bool bok[4];
            #pragma unroll
            for (int t = 0; t < 4; ++t) {
                int iy, ix; bool ok = nok[t];
                if (mode == 0) {                       // fwd stride2, pad_lo=0
                    iy = oyv[t] * 2 + ky; ix = oxv[t] * 2 + kx;
                    ok = ok && (iy < Hin) && (ix < Win);
                } else if (mode == 1) {                // 1x1
                    iy = oyv[t]; ix = oxv[t];
                } else {                               // transpose stride2, pad_a=2
                    int dy = oyv[t] + ky - 2, dx = oxv[t] + kx - 2;
                    ok = ok && (dy >= 0) && (dx >= 0) && !(dy & 1) && !(dx & 1);
                    iy = dy >> 1; ix = dx >> 1;
                    ok = ok && (iy < Hin) && (ix < Win);
                }
                bok[t] = ok;
                bbase[t] = in + (size_t)bbv[t] * Cin * HWin + (size_t)iy * Win + ix;
                if (ok) __builtin_prefetch(bbase[t], 0, 1);   // global_prefetch_b8
            }

            for (int c0 = 0; c0 < Cin; c0 += 32) {
                // A fragment (shared by all 4 column tiles)
                v16h af;
                #pragma unroll
                for (int e = 0; e < 16; ++e) {
                    int ci = c0 + aKbase + e;
                    float av = (m_ok && ci < Cin) ? wrow[ci * KHW + kyx] : 0.0f;
                    af[e] = (_Float16)av;
                }
                #pragma unroll
                for (int t = 0; t < 4; ++t) {
                    v16h bf;
                    #pragma unroll
                    for (int e = 0; e < 16; ++e) {
                        int ci = c0 + bKbase + e;
                        float bv = (bok[t] && ci < Cin)
                                 ? bbase[t][(size_t)ci * HWin] : 0.0f;
                        bf[e] = (_Float16)bv;
                    }
                    acc[t] = __builtin_amdgcn_wmma_f32_16x16x32_f16(
                                 false, af, false, bf, (short)0, acc[t],
                                 false, false);
                }
            }
        }
    }

    // epilogue: bias (+ReLU) and store
    float bv8[8];
    #pragma unroll
    for (int vv = 0; vv < 8; ++vv) {
        int m = tileM * 16 + vv + ((lane & 16) ? 8 : 0);
        bv8[vv] = (m < Cout) ? bias[m] : 0.0f;
    }
    #pragma unroll
    for (int t = 0; t < 4; ++t) {
        #pragma unroll
        for (int vv = 0; vv < 8; ++vv) {
            int m = tileM * 16 + vv + ((lane & 16) ? 8 : 0);
            if (m < Cout && nok[t]) {
                float r = acc[t][vv] + bv8[vv];
                if (relu) r = fmaxf(r, 0.0f);
                out[((bbv[t] * Cout + m) * Hout + oyv[t]) * Wout + oxv[t]] = r;
            }
        }
    }
}

// ---------------------------------------------------------------------------
// Saliency: one block (256 thr) per sample. zmap layout (b, d=256, n=1024).
// ---------------------------------------------------------------------------
__global__ void saliency_kernel(const float* __restrict__ zmap,
                                float* __restrict__ mask)
{
    const int b = blockIdx.x;
    const int t = threadIdx.x;            // 0..255, doubles as dim index
    __shared__ float bgn[256];
    __shared__ float red[256];

    const float* zb = zmap + (size_t)b * 256 * 1024;

    // bg[d] = sum_{n in border} z[n][d] / 348, then l2-normalize over d
    float s = 0.0f;
    for (int n = 0; n < 1024; ++n) {
        int y = n >> 5, x = n & 31;
        if ((y < 3) || (y >= 29) || (x < 3) || (x >= 29))
            s += zb[t * 1024 + n];
    }
    s *= (1.0f / 348.0f);
    red[t] = s * s; __syncthreads();
    for (int o = 128; o > 0; o >>= 1) { if (t < o) red[t] += red[t + o]; __syncthreads(); }
    float bnorm = fmaxf(sqrtf(red[0]), 1e-12f); __syncthreads();
    bgn[t] = s / bnorm;
    __syncthreads();

    float sim[4];
    float lmn = 1e30f, lmx = -1e30f;
    for (int j = 0; j < 4; ++j) {
        int n = j * 256 + t;
        float dot = 0.0f, ss = 0.0f;
        for (int d = 0; d < 256; ++d) {
            float zv = zb[d * 1024 + n];
            dot += zv * bgn[d];
            ss  += zv * zv;
        }
        float sv = dot / fmaxf(sqrtf(ss), 1e-12f);
        sim[j] = sv;
        lmn = fminf(lmn, sv); lmx = fmaxf(lmx, sv);
    }
    red[t] = lmn; __syncthreads();
    for (int o = 128; o > 0; o >>= 1) { if (t < o) red[t] = fminf(red[t], red[t + o]); __syncthreads(); }
    float mn = red[0]; __syncthreads();
    red[t] = lmx; __syncthreads();
    for (int o = 128; o > 0; o >>= 1) { if (t < o) red[t] = fmaxf(red[t], red[t + o]); __syncthreads(); }
    float mx = red[0]; __syncthreads();

    float inv = 1.0f / (mx - mn);
    for (int j = 0; j < 4; ++j) {
        int n = j * 256 + t;
        mask[b * 1024 + n] = 1.0f - (sim[j] - mn) * inv;
    }
}

// ---------------------------------------------------------------------------
// Cluster + sinkhorn: one block (256 thr) per sample. Sinkhorn done in the
// equivalent diagonal-scaling form Q = diag(u) E^T diag(v):
//   row pass:  u[k] = 1 / (32   * sum_n E[n][k] * v[n])
//   col pass:  v[n] = 1 / (1024 * sum_k u[k] * E[n][k])
// q[n][k] = 1024 * u[k] * E[n][k] * v[n].
// ---------------------------------------------------------------------------
__global__ void cluster_kernel(const float* __restrict__ zmap,
                               const float* __restrict__ proto_init,
                               float* __restrict__ Ews,
                               float* __restrict__ logits,
                               float* __restrict__ idxmap)
{
    const int b = blockIdx.x;
    const int t = threadIdx.x;           // 0..255, doubles as dim index
    __shared__ float proto[32 * 256];    // 32 KB
    __shared__ float vvec[1024];
    __shared__ float red[256];
    __shared__ float uvec[32];
    __shared__ float Sk[32];
    __shared__ float cnts[32];
    __shared__ int   idxs[1024];

    const float* zb = zmap + (size_t)b * 256 * 1024;
    float* Eb = Ews + (size_t)b * 1024 * 32;

    // protos = l2n(proto_init[b])
    for (int k = 0; k < 32; ++k) {
        float pv = proto_init[((size_t)b * 32 + k) * 256 + t];
        red[t] = pv * pv; __syncthreads();
        for (int o = 128; o > 0; o >>= 1) { if (t < o) red[t] += red[t + o]; __syncthreads(); }
        float nrm = fmaxf(sqrtf(red[0]), 1e-12f); __syncthreads();
        proto[k * 256 + t] = pv / nrm;
    }
    __syncthreads();

    for (int it = 0; it < 3; ++it) {
        // ---- s = softmax_k(zn . proto_k); E = exp(s / 0.05) ----
        for (int j = 0; j < 4; ++j) {
            int n = j * 256 + t;
            float dacc[32];
            #pragma unroll
            for (int k = 0; k < 32; ++k) dacc[k] = 0.0f;
            float ss = 0.0f;
            for (int d0 = 0; d0 < 256; d0 += 8) {
                float zz[8];
                #pragma unroll
                for (int dd = 0; dd < 8; ++dd) {
                    zz[dd] = zb[(d0 + dd) * 1024 + n];
                    ss += zz[dd] * zz[dd];
                }
                #pragma unroll
                for (int k = 0; k < 32; ++k) {
                    float a = 0.0f;
                    #pragma unroll
                    for (int dd = 0; dd < 8; ++dd)
                        a += zz[dd] * proto[k * 256 + d0 + dd];
                    dacc[k] += a;
                }
            }
            float rinv = 1.0f / fmaxf(sqrtf(ss), 1e-12f);
            float mxv = -1e30f;
            #pragma unroll
            for (int k = 0; k < 32; ++k) { dacc[k] *= rinv; mxv = fmaxf(mxv, dacc[k]); }
            float sum = 0.0f;
            #pragma unroll
            for (int k = 0; k < 32; ++k) { dacc[k] = expf(dacc[k] - mxv); sum += dacc[k]; }
            float sinv = 1.0f / sum;
            #pragma unroll
            for (int k = 0; k < 32; ++k)
                Eb[n * 32 + k] = expf(dacc[k] * sinv * 20.0f);   // 1/EPS = 20
        }
        for (int j = 0; j < 4; ++j) vvec[j * 256 + t] = 1.0f;
        __syncthreads();

        // ---- 3 sinkhorn iterations ----
        for (int si = 0; si < 3; ++si) {
            if (t < 32) Sk[t] = 0.0f;
            __syncthreads();
            float sl[32];
            #pragma unroll
            for (int k = 0; k < 32; ++k) sl[k] = 0.0f;
            for (int j = 0; j < 4; ++j) {
                int n = j * 256 + t;
                float vn = vvec[n];
                #pragma unroll
                for (int k = 0; k < 32; ++k)
                    sl[k] += Eb[n * 32 + k] * vn;
            }
            #pragma unroll
            for (int k = 0; k < 32; ++k) atomicAdd(&Sk[k], sl[k]);
            __syncthreads();
            if (t < 32) uvec[t] = 1.0f / (32.0f * Sk[t]);
            __syncthreads();
            for (int j = 0; j < 4; ++j) {
                int n = j * 256 + t;
                float Tn = 0.0f;
                #pragma unroll
                for (int k = 0; k < 32; ++k) Tn += uvec[k] * Eb[n * 32 + k];
                vvec[n] = 1.0f / (1024.0f * Tn);
            }
            __syncthreads();
        }

        // ---- q, argmax (logits + idx written on final iter) ----
        for (int j = 0; j < 4; ++j) {
            int n = j * 256 + t;
            float vn = vvec[n] * 1024.0f;
            int best = 0; float bestv = -1e30f;
            #pragma unroll
            for (int k = 0; k < 32; ++k) {
                float q = uvec[k] * Eb[n * 32 + k] * vn;
                if (it == 2) logits[((size_t)b * 32 + k) * 1024 + n] = q;
                if (q > bestv) { bestv = q; best = k; }
            }
            idxs[n] = best;
            if (it == 2) idxmap[b * 1024 + n] = (float)best;
        }
        __syncthreads();

        // ---- proto update (skipped after last sinkhorn) ----
        if (it < 2) {
            if (t < 32) cnts[t] = 0.0f;
            __syncthreads();
            for (int j = 0; j < 4; ++j) atomicAdd(&cnts[idxs[j * 256 + t]], 1.0f);
            for (int k = 0; k < 32; ++k) proto[k * 256 + t] = 0.0f;
            __syncthreads();
            // thread t owns dim d=t: conflict-free LDS accumulation, raw z (not zn)
            for (int n = 0; n < 1024; ++n) {
                float zv = zb[t * 1024 + n];
                proto[idxs[n] * 256 + t] += zv;
            }
            __syncthreads();
            for (int k = 0; k < 32; ++k) {
                float cc = cnts[k];
                float nv = (cc > 0.0f) ? proto[k * 256 + t] / cc : 0.0f; // nan_to_num
                red[t] = nv * nv; __syncthreads();
                for (int o = 128; o > 0; o >>= 1) { if (t < o) red[t] += red[t + o]; __syncthreads(); }
                float nrm = fmaxf(sqrtf(red[0]), 1e-12f); __syncthreads();
                proto[k * 256 + t] = nv / nrm;
                __syncthreads();
            }
        }
        __syncthreads();
    }
}

// ---------------------------------------------------------------------------
static void launch_conv(const float* in, const float* w, const float* bias, float* out,
                        int B, int Cin, int Hin, int Win,
                        int Cout, int Hout, int Wout,
                        int KH, int KW, int mode, int relu, hipStream_t s)
{
    int Ntot = B * Hout * Wout;
    int nstrips = (Ntot + 63) / 64;
    dim3 grid((nstrips + 7) / 8, (Cout + 15) / 16);
    conv_wmma_kernel<<<grid, 256, 0, s>>>(in, w, bias, out,
                                          B, Cin, Hin, Win, Cout, Hout, Wout,
                                          KH, KW, mode, relu);
}

extern "C" void kernel_launch(void* const* d_in, const int* in_sizes, int n_in,
                              void* d_out, int out_size, void* d_ws, size_t ws_size,
                              hipStream_t stream)
{
    const float* x   = (const float*)d_in[0];
    const float* pin = (const float*)d_in[1];
    const float* ew1 = (const float*)d_in[2];  const float* eb1 = (const float*)d_in[3];
    const float* ew2 = (const float*)d_in[4];  const float* eb2 = (const float*)d_in[5];
    const float* ew3 = (const float*)d_in[6];  const float* eb3 = (const float*)d_in[7];
    const float* ew4 = (const float*)d_in[8];  const float* eb4 = (const float*)d_in[9];
    const float* dw1 = (const float*)d_in[10]; const float* db1 = (const float*)d_in[11];
    const float* dw2 = (const float*)d_in[12]; const float* db2 = (const float*)d_in[13];
    const float* dw3 = (const float*)d_in[14]; const float* db3 = (const float*)d_in[15];
    const float* dw4 = (const float*)d_in[16]; const float* db4 = (const float*)d_in[17];

    // d_out regions (return order): x_bar, zmap, mask, logits_map, idx_map
    float* out     = (float*)d_out;
    float* xbar    = out;                       // 16*3*256*256   = 3,145,728
    float* zmap    = out + 3145728;             // 16*256*32*32   = 4,194,304
    float* maskO   = zmap + 4194304;            // 16*1024        =    16,384
    float* logitsO = maskO + 16384;             // 16*32*1024     =   524,288
    float* idxO    = logitsO + 524288;          // 16*32*32       =    16,384

    // workspace: buf64 (h1/d3, 64MB) | buf32 (h2/d2, 32MB) | buf16 (h3/d1, 16MB) | E (2MB)
    char*  ws    = (char*)d_ws;
    float* buf64 = (float*)ws;
    float* buf32 = (float*)(ws + 67108864);
    float* buf16 = (float*)(ws + 67108864 + 33554432);
    float* Ews   = (float*)(ws + 67108864 + 33554432 + 16777216);   // needs ~114MB total

    // ---- encoder ----
    launch_conv(x,     ew1, eb1, buf64, 16,   3, 256, 256,  64, 128, 128, 3, 3, 0, 1, stream);
    launch_conv(buf64, ew2, eb2, buf32, 16,  64, 128, 128, 128,  64,  64, 3, 3, 0, 1, stream);
    launch_conv(buf32, ew3, eb3, buf16, 16, 128,  64,  64, 256,  32,  32, 3, 3, 0, 1, stream);
    launch_conv(buf16, ew4, eb4, zmap,  16, 256,  32,  32, 256,  32,  32, 1, 1, 1, 0, stream);
    // ---- decoder ----
    launch_conv(zmap,  dw1, db1, buf16, 16, 256,  32,  32, 256,  32,  32, 1, 1, 1, 1, stream);
    launch_conv(buf16, dw2, db2, buf32, 16, 256,  32,  32, 128,  64,  64, 3, 3, 2, 1, stream);
    launch_conv(buf32, dw3, db3, buf64, 16, 128,  64,  64,  64, 128, 128, 3, 3, 2, 1, stream);
    launch_conv(buf64, dw4, db4, xbar,  16,  64, 128, 128,   3, 256, 256, 3, 3, 2, 0, stream);
    // ---- saliency + clustering ----
    saliency_kernel<<<16, 256, 0, stream>>>(zmap, maskO);
    cluster_kernel<<<16, 256, 0, stream>>>(zmap, pin, Ews, logitsO, idxO);
}
